// GRUCell_87265145520378
// MI455X (gfx1250) — matrix operand order repaired and checked by reading, not compiled, not run on hardware
//
#include <hip/hip_runtime.h>
#include <cstdint>
#include <cstddef>

namespace {

constexpr int kH     = 1024;          // INPUT_SIZE == HIDDEN_SIZE
constexpr int kBatch = 8192;
constexpr int kKT    = kH / 32;       // 32  K-tiles (WMMA K-chunk = 32)
constexpr int kMT    = kBatch / 16;   // 512 M-tiles
constexpr int kNT    = kH / 16;       // 64  N-tiles per logical gate
constexpr int kFrag  = 12;            // B fragments per (nt, kt)
constexpr int kStageBytes = kFrag * 1024;  // 12 KB staged per K-step

typedef __attribute__((ext_vector_type(16))) __bf16 v16bf;
typedef __attribute__((ext_vector_type(8)))  float  v8f;

__device__ __forceinline__ unsigned short f2bf(float x) {
  unsigned int u = __float_as_uint(x);
  u += 0x7fffu + ((u >> 16) & 1u);              // round-to-nearest-even
  return (unsigned short)(u >> 16);
}
__device__ __forceinline__ float bf2f(unsigned short h) {
  return __uint_as_float((unsigned int)h << 16);
}

__device__ __forceinline__ v8f wmma_bf16(v16bf a, v16bf b, v8f c) {
  return __builtin_amdgcn_wmma_f32_16x16x32_bf16(false, a, false, b, (short)0, c,
                                                 false, false);
}

// Pack activation matrix X[M x 1024] (row-major f32) into WMMA A-fragment
// order (16x32 bf16 tiles), split into bf16 hi + lo for error compensation.
// A layout (ISA 7.12.2): lanes 0-15 hold row M=lane, K = {0..7,16..23};
// lanes 16-31 hold row M=lane-16, K = {8..15,24..31}.
__global__ void pack_a_kernel(const float* __restrict__ X,
                              unsigned short* __restrict__ hi,
                              unsigned short* __restrict__ lo) {
  int t    = blockIdx.x * blockDim.x + threadIdx.x;
  int lane = t & 31;
  int tile = t >> 5;                      // tile = mt * kKT + kt
  if (tile >= kMT * kKT) return;
  int kt = tile & (kKT - 1);
  int mt = tile >> 5;
  int m  = mt * 16 + (lane & 15);
  int kb = kt * 32 + ((lane >= 16) ? 8 : 0);
  size_t o = (size_t)tile * 512 + (size_t)lane * 16;
#pragma unroll
  for (int e = 0; e < 16; ++e) {
    int k = kb + e + ((e >= 8) ? 8 : 0);
    float x = X[(size_t)m * kH + k];
    unsigned short h = f2bf(x);
    hi[o + e] = h;
    lo[o + e] = f2bf(x - bf2f(h));
  }
}

// Pack one gate-half of a weight matrix W[1024 x N] into the combined B
// buffer Bcomb laid out as [nt][kt][frag][1KB], frag pair (fhi, fhi+1) =
// (hi, lo). B-fragment order (ISA 7.12.2): lanes 0-15 = column N=lane,
// K={0..7,16..23}; lanes 16-31 = column N=lane-16, K={8..15,24..31}.
__global__ void pack_b_comb_kernel(const float* __restrict__ W, int N, int colOff,
                                   int fhi, unsigned short* __restrict__ Bcomb) {
  int t    = blockIdx.x * blockDim.x + threadIdx.x;
  int lane = t & 31;
  int tile = t >> 5;                      // tile = nt * kKT + kt, nt in 0..63
  if (tile >= kNT * kKT) return;
  int kt = tile & (kKT - 1);
  int nt = tile >> 5;
  int n  = colOff + nt * 16 + (lane & 15);
  int kb = kt * 32 + ((lane >= 16) ? 8 : 0);
  size_t baseHi = ((size_t)tile * kFrag + fhi) * 512;   // ushort units
  size_t baseLo = baseHi + 512;
  size_t o = (size_t)lane * 16;
#pragma unroll
  for (int e = 0; e < 16; ++e) {
    float x = W[(size_t)(kb + e + ((e >= 8) ? 8 : 0)) * N + n];
    unsigned short h = f2bf(x);
    Bcomb[baseHi + o + e] = h;
    Bcomb[baseLo + o + e] = f2bf(x - bf2f(h));
  }
}

// Fused GRU cell. One wave = one 16x16 output tile with four accumulators
// (z, r, input@W_c, hx@W_hc). B fragments are staged per-block into LDS with
// CDNA5 async-to-LDS copies (double buffered), shared by 8 waves; A streams
// from global. 18 v_wmma per K-step per wave.
// Frag order: 0 bizH 1 bizL 2 bhzH 3 bhzL 4 birH 5 birL 6 bhrH 7 bhrL
//             8 bcH  9 bcL 10 bhcH 11 bhcL
__global__ __launch_bounds__(256) void gru_wmma_kernel(
    const unsigned short* __restrict__ AinH, const unsigned short* __restrict__ AinL,
    const unsigned short* __restrict__ AhxH, const unsigned short* __restrict__ AhxL,
    const unsigned short* __restrict__ Bcomb,
    const float* __restrict__ b_ih, const float* __restrict__ b_hh,
    const float* __restrict__ b_c,  const float* __restrict__ b_hc,
    const float* __restrict__ hx,   float* __restrict__ out) {
  __shared__ __align__(32) unsigned char smem[2 * kStageBytes];

  const int tid  = threadIdx.x;
  const int lane = tid & 31;
  const int wave = tid >> 5;
  const int nt   = blockIdx.x;              // 0..63  (output N-tile)
  const int mt   = blockIdx.y * 8 + wave;   // 0..511 (output M-tile)

  const size_t aoff = (size_t)mt * kKT * 32 + lane;
  const v16bf* ainH = (const v16bf*)AinH + aoff;
  const v16bf* ainL = (const v16bf*)AinL + aoff;
  const v16bf* ahxH = (const v16bf*)AhxH + aoff;
  const v16bf* ahxL = (const v16bf*)AhxL + aoff;

  const unsigned long long gB0 =
      (unsigned long long)(uintptr_t)Bcomb + (unsigned long long)nt * kKT * kStageBytes;
  const unsigned int lB0 = (unsigned int)(uintptr_t)(&smem[0]);  // low 32 bits = LDS offset

  // Stage the 12 KB of B fragments for one K-step: 256 threads x 3 x 16B.
  auto stage = [&](int kt, int buf) {
    const unsigned long long g = gB0 + (unsigned long long)kt * kStageBytes;
    const unsigned int l = lB0 + (unsigned)buf * kStageBytes;
#pragma unroll
    for (int i = 0; i < 3; ++i) {
      const unsigned int off = ((unsigned)tid + 256u * i) * 16u;
      const unsigned long long ga = g + off;
      const unsigned int la = l + off;
      asm volatile("global_load_async_to_lds_b128 %0, %1, off"
                   :: "v"(la), "v"(ga) : "memory");
    }
  };

  stage(0, 0);

  v8f az = {}, ar = {}, ac = {}, ah = {};

#pragma unroll 2
  for (int kt = 0; kt < kKT; ++kt) {
    const int cur = kt & 1;
    asm volatile("s_wait_asynccnt 0x0" ::: "memory");  // my stage writes landed
    __syncthreads();                                    // everyone's landed + prev reads done
    if (kt + 1 < kKT) stage(kt + 1, cur ^ 1);

    const size_t s = (size_t)kt * 32;
    const v16bf aiH = ainH[s], aiL = ainL[s];
    const v16bf axH = ahxH[s], axL = ahxL[s];
    const v16bf* bb = (const v16bf*)(smem + cur * kStageBytes) + lane;

    v16bf b;
    b = bb[0 * 32];  az = wmma_bf16(aiH, b, az); az = wmma_bf16(aiL, b, az);
    b = bb[1 * 32];  az = wmma_bf16(aiH, b, az);
    b = bb[2 * 32];  az = wmma_bf16(axH, b, az); az = wmma_bf16(axL, b, az);
    b = bb[3 * 32];  az = wmma_bf16(axH, b, az);
    b = bb[4 * 32];  ar = wmma_bf16(aiH, b, ar); ar = wmma_bf16(aiL, b, ar);
    b = bb[5 * 32];  ar = wmma_bf16(aiH, b, ar);
    b = bb[6 * 32];  ar = wmma_bf16(axH, b, ar); ar = wmma_bf16(axL, b, ar);
    b = bb[7 * 32];  ar = wmma_bf16(axH, b, ar);
    b = bb[8 * 32];  ac = wmma_bf16(aiH, b, ac); ac = wmma_bf16(aiL, b, ac);
    b = bb[9 * 32];  ac = wmma_bf16(aiH, b, ac);
    b = bb[10 * 32]; ah = wmma_bf16(axH, b, ah); ah = wmma_bf16(axL, b, ah);
    b = bb[11 * 32]; ah = wmma_bf16(axH, b, ah);
  }

  // Epilogue. C/D layout: VGPR r -> row M = r (+8 for lanes 16-31), col = lane%16.
  const int n    = nt * 16 + (lane & 15);
  const int mrow = mt * 16 + (lane >> 4) * 8;
  const float bz  = b_ih[n] + b_hh[n];
  const float br  = b_ih[kH + n] + b_hh[kH + n];
  const float bcv = b_c[n];
  const float bhv = b_hc[n];
#pragma unroll
  for (int r = 0; r < 8; ++r) {
    const int m = mrow + r;
    const float hprev = hx[(size_t)m * kH + n];
    const float z  = 1.0f / (1.0f + __expf(-(az[r] + bz)));
    const float rr = 1.0f / (1.0f + __expf(-(ar[r] + br)));
    const float cand = tanhf(ac[r] + bcv + rr * (ah[r] + bhv));
    out[(size_t)m * kH + n] = (1.0f - z) * hprev + z * cand;
  }
}

} // namespace

extern "C" void kernel_launch(void* const* d_in, const int* in_sizes, int n_in,
                              void* d_out, int out_size, void* d_ws, size_t ws_size,
                              hipStream_t stream) {
  (void)in_sizes; (void)n_in; (void)out_size; (void)ws_size;
  const float* input = (const float*)d_in[0];
  const float* hx    = (const float*)d_in[1];
  const float* W_ih  = (const float*)d_in[2];
  const float* b_ih  = (const float*)d_in[3];
  const float* W_hh  = (const float*)d_in[4];
  const float* b_hh  = (const float*)d_in[5];
  const float* W_c   = (const float*)d_in[6];
  const float* b_c   = (const float*)d_in[7];
  const float* W_hc  = (const float*)d_in[8];
  const float* b_hc  = (const float*)d_in[9];
  float* out = (float*)d_out;

  const size_t MK    = (size_t)kBatch * kH;                 // 8 Mi elems
  const size_t BCOMB = (size_t)kNT * kKT * kFrag * 512;     // 12 Mi elems (24 MiB)

  unsigned short* p = (unsigned short*)d_ws;  // 88 MiB total
  unsigned short* AinH  = p; p += MK;
  unsigned short* AinL  = p; p += MK;
  unsigned short* AhxH  = p; p += MK;
  unsigned short* AhxL  = p; p += MK;
  unsigned short* Bcomb = p; p += BCOMB;

  const int TPB = 256;
  const int aThreads = kMT * kKT * 32;   // 524288
  const int bThreads = kNT * kKT * 32;   // 65536 per gate-half

  pack_a_kernel<<<aThreads / TPB, TPB, 0, stream>>>(input, AinH, AinL);
  pack_a_kernel<<<aThreads / TPB, TPB, 0, stream>>>(hx,    AhxH, AhxL);
  pack_b_comb_kernel<<<bThreads / TPB, TPB, 0, stream>>>(W_ih, 2 * kH, 0,  0,  Bcomb);
  pack_b_comb_kernel<<<bThreads / TPB, TPB, 0, stream>>>(W_hh, 2 * kH, 0,  2,  Bcomb);
  pack_b_comb_kernel<<<bThreads / TPB, TPB, 0, stream>>>(W_ih, 2 * kH, kH, 4,  Bcomb);
  pack_b_comb_kernel<<<bThreads / TPB, TPB, 0, stream>>>(W_hh, 2 * kH, kH, 6,  Bcomb);
  pack_b_comb_kernel<<<bThreads / TPB, TPB, 0, stream>>>(W_c,  kH,     0,  8,  Bcomb);
  pack_b_comb_kernel<<<bThreads / TPB, TPB, 0, stream>>>(W_hc, kH,     0,  10, Bcomb);

  dim3 grid(kNT, kMT / 8);  // (64, 64), 8 waves/block = 8 M-tiles per N-tile
  gru_wmma_kernel<<<grid, 256, 0, stream>>>(AinH, AinL, AhxH, AhxL, Bcomb,
                                            b_ih, b_hh, b_c, b_hc, hx, out);
}
